// HeteroGATEncoder_17781164606101
// MI455X (gfx1250) — matrix operand order repaired and checked by reading, not compiled, not run on hardware
//
#include <hip/hip_runtime.h>
#include <hip/hip_bf16.h>
#include <math.h>

typedef __attribute__((ext_vector_type(16))) __bf16 v16bf;
typedef __attribute__((ext_vector_type(8)))  float  v8f;

#define NU 50000
#define NI 50000
#define NE 800000
#define NEG_SLOPE 0.2f

// ---------------- helpers ----------------
static __device__ __forceinline__ unsigned short f2bf(float f) {
  unsigned int u = __float_as_uint(f);
  unsigned int r = u + 0x7FFFu + ((u >> 16) & 1u);   // round-to-nearest-even
  return (unsigned short)(r >> 16);
}
static __device__ __forceinline__ float bf2f(unsigned short h) {
  return __uint_as_float(((unsigned int)h) << 16);
}
static __device__ __forceinline__ void atomicMaxF(float* addr, float val) {
  // ordered-int trick; works for all non-NaN floats when initialized to -1e30
  if (val >= 0.f) atomicMax((int*)addr, __float_as_int(val));
  else            atomicMin((unsigned int*)addr, __float_as_uint(val));
}

// ---------------- conversion / small kernels ----------------
__global__ void k_f32_to_bf16(const float* __restrict__ in, unsigned short* __restrict__ out, int n) {
  int i = blockIdx.x * blockDim.x + threadIdx.x;
  if (i < n) out[i] = f2bf(in[i]);
}
// Wt[n*K+k] = W[k*N+n]  (store B transposed, bf16)
__global__ void k_transpose_bf16(const float* __restrict__ W, unsigned short* __restrict__ Wt, int K, int N) {
  int i = blockIdx.x * blockDim.x + threadIdx.x;
  if (i < K * N) { int n = i / K, k = i % K; Wt[i] = f2bf(W[k * N + n]); }
}
__global__ void k_fill_f32(float* __restrict__ p, float v, int n) {
  int i = blockIdx.x * blockDim.x + threadIdx.x;
  if (i < n) p[i] = v;
}
// u[k*H+h] = sum_c W[k*(H*C)+h*C+c] * a[h*C+c]   (attention vector folded into weights)
__global__ void k_make_u(const float* __restrict__ W, const float* __restrict__ a,
                         float* __restrict__ u, int K, int H, int C) {
  int i = blockIdx.x * blockDim.x + threadIdx.x;
  if (i >= K * H) return;
  int k = i / H, h = i % H;
  float s = 0.f;
  for (int c = 0; c < C; ++c) s += W[k * (H * C) + h * C + c] * a[h * C + c];
  u[i] = s;
}
// al[n*H+h] = sum_k bf(A[n*K+k]) * u[k*H+h]
__global__ void k_al(const unsigned short* __restrict__ A, const float* __restrict__ u,
                     float* __restrict__ al, int Nn, int K, int H) {
  int i = blockIdx.x * blockDim.x + threadIdx.x;
  if (i >= Nn * H) return;
  int n = i / H, h = i % H;
  float s = 0.f;
  for (int k = 0; k < K; ++k) s += bf2f(A[n * K + k]) * u[k * H + h];
  al[i] = s;
}

// ---------------- WMMA GEMM: C[M,N] = A[M,K] * B[K,N], A bf16 row-major, Bt = B^T bf16 [N,K]
template <int NT>
__global__ __launch_bounds__(256) void k_wmma_gemm(
    const unsigned short* __restrict__ A,
    const unsigned short* __restrict__ Bt,
    const float* __restrict__ bias,          // [N] or nullptr
    float* __restrict__ Cf,                  // f32 out or nullptr
    unsigned short* __restrict__ Cbf,        // bf16 out or nullptr
    int M, int N, int K, int act)            // act: 1 = ELU
{
  const int lane  = threadIdx.x & 31;
  const int wave  = blockIdx.x * (blockDim.x >> 5) + (threadIdx.x >> 5);
  const int tilesN = N / (16 * NT);
  const int tiles  = (M / 16) * tilesN;
  if (wave >= tiles) return;
  const int tm = wave / tilesN, tn = wave % tilesN;
  const int m0 = tm * 16, n0 = tn * 16 * NT;
  const int half = lane >> 4, l15 = lane & 15;

  v8f acc[NT] = {};
  for (int kk = 0; kk < K; kk += 32) {
    // A fragment: 16x32 bf16 (lane l15 = row M, K pairs per ISA layout)
    union { v16bf v; unsigned int u[8]; } a;
    const unsigned short* Ap = A + (size_t)(m0 + l15) * K + kk;
#pragma unroll
    for (int j = 0; j < 8; ++j) {
      int k = ((j >> 2) << 4) + half * 8 + ((j & 3) << 1);
      a.u[j] = *(const unsigned int*)(Ap + k);
    }
#pragma unroll
    for (int t = 0; t < NT; ++t) {
      // B fragment: 32x16 bf16 (lane l15 = col N, K pairs per ISA layout)
      union { v16bf v; unsigned int u[8]; } b;
      const unsigned short* Bp = Bt + (size_t)(n0 + t * 16 + l15) * K + kk;
#pragma unroll
      for (int j = 0; j < 8; ++j) b.u[j] = *(const unsigned int*)(Bp + ((j >> 2) << 4) + half * 8 + ((j & 3) << 1));
      acc[t] = __builtin_amdgcn_wmma_f32_16x16x32_bf16(
          false, a.v, false, b.v, (short)0, acc[t], false, false);
    }
  }
  // epilogue: lane holds N=l15, rows M = half*8 + r
#pragma unroll
  for (int t = 0; t < NT; ++t) {
    int col = n0 + t * 16 + l15;
    float bv = bias ? bias[col] : 0.f;
#pragma unroll
    for (int r = 0; r < 8; ++r) {
      int row = m0 + half * 8 + r;
      float v = acc[t][r] + bv;
      if (act) v = v > 0.f ? v : (expf(v) - 1.f);
      if (Cf)  Cf[(size_t)row * N + col] = v;
      if (Cbf) Cbf[(size_t)row * N + col] = f2bf(v);
    }
  }
}

// ---------------- edge-phase kernels ----------------
__global__ void k_alpha_max(const int* __restrict__ src, const int* __restrict__ dst,
                            const float* __restrict__ als, const float* __restrict__ ald,
                            float* __restrict__ alpha, float* __restrict__ m, int E, int H) {
  int t = blockIdx.x * blockDim.x + threadIdx.x;
  if (t >= E * H) return;
  int e = t / H, h = t % H;
  int s = src[e], d = dst[e];
  float a = als[s * H + h] + ald[d * H + h];
  a = a > 0.f ? a : NEG_SLOPE * a;
  alpha[t] = a;
  atomicMaxF(&m[d * H + h], a);
}
__global__ void k_exp_sum(const int* __restrict__ dst, const float* __restrict__ alpha,
                          const float* __restrict__ m, float* __restrict__ ealpha,
                          float* __restrict__ denom, int E, int H) {
  int t = blockIdx.x * blockDim.x + threadIdx.x;
  if (t >= E * H) return;
  int e = t / H, h = t % H;
  int d = dst[e];
  float ea = expf(alpha[t] - m[d * H + h]);
  ealpha[t] = ea;
  atomicAdd(&denom[d * H + h], ea);
}
__global__ void k_scatter(const int* __restrict__ src, const int* __restrict__ dst,
                          const float* __restrict__ hs, const float* __restrict__ ealpha,
                          const float* __restrict__ denom, float* __restrict__ acc,
                          int E, int H, int C) {
  int C4 = C >> 2;
  int t = blockIdx.x * blockDim.x + threadIdx.x;
  if (t >= E * H * C4) return;
  int c4 = t % C4, eh = t / C4;
  int h = eh % H, e = eh / H;
  int s = src[e], d = dst[e];
  float coef = ealpha[e * H + h] / (denom[d * H + h] + 1e-16f);
  const float4 v = *(const float4*)(hs + (size_t)s * H * C + h * C + c4 * 4);
  float* o = acc + (size_t)d * H * C + h * C + c4 * 4;
  atomicAdd(o + 0, v.x * coef);
  atomicAdd(o + 1, v.y * coef);
  atomicAdd(o + 2, v.z * coef);
  atomicAdd(o + 3, v.w * coef);
}
__global__ void k_finish_elu_bf16(const float* __restrict__ acc, const float* __restrict__ b,
                                  unsigned short* __restrict__ out, int n, int F) {
  int i = blockIdx.x * blockDim.x + threadIdx.x;
  if (i >= n) return;
  float v = acc[i] + b[i % F];
  v = v > 0.f ? v : (expf(v) - 1.f);
  out[i] = f2bf(v);
}
__global__ void k_add_bias(float* __restrict__ o, const float* __restrict__ b, int n, int F) {
  int i = blockIdx.x * blockDim.x + threadIdx.x;
  if (i < n) o[i] += b[i % F];
}

// ---------------- host orchestration ----------------
extern "C" void kernel_launch(void* const* d_in, const int* in_sizes, int n_in,
                              void* d_out, int out_size, void* d_ws, size_t ws_size,
                              hipStream_t stream) {
  const float* x_user = (const float*)d_in[0];
  const float* x_item = (const float*)d_in[1];
  const int*   e_u2i  = (const int*)d_in[2];
  const int*   e_i2u  = (const int*)d_in[3];
  const float* p_user_w = (const float*)d_in[4];
  const float* p_user_b = (const float*)d_in[5];
  const float* p_item_w = (const float*)d_in[6];
  const float* p_item_b = (const float*)d_in[7];
  const float *l0u_ws = (const float*)d_in[8],  *l0u_wd = (const float*)d_in[9];
  const float *l0u_as = (const float*)d_in[10], *l0u_ad = (const float*)d_in[11], *l0u_b = (const float*)d_in[12];
  const float *l0i_ws = (const float*)d_in[13], *l0i_wd = (const float*)d_in[14];
  const float *l0i_as = (const float*)d_in[15], *l0i_ad = (const float*)d_in[16], *l0i_b = (const float*)d_in[17];
  const float *l1u_ws = (const float*)d_in[18], *l1u_wd = (const float*)d_in[19];
  const float *l1u_as = (const float*)d_in[20], *l1u_ad = (const float*)d_in[21], *l1u_b = (const float*)d_in[22];
  const float *l1i_ws = (const float*)d_in[23], *l1i_wd = (const float*)d_in[24];
  const float *l1i_as = (const float*)d_in[25], *l1i_ad = (const float*)d_in[26], *l1i_b = (const float*)d_in[27];

  // bump allocator over workspace
  char* base = (char*)d_ws;
  size_t off = 0;
  auto alloc = [&](size_t bytes) -> void* {
    void* p = base + off;
    off = (off + bytes + 255) & ~(size_t)255;
    return p;
  };
  unsigned short* us_xu  = (unsigned short*)alloc((size_t)NU * 128 * 2);
  unsigned short* us_xi  = (unsigned short*)alloc((size_t)NI * 128 * 2);
  unsigned short* us_hu  = (unsigned short*)alloc((size_t)NU * 64 * 2);
  unsigned short* us_hi  = (unsigned short*)alloc((size_t)NI * 64 * 2);
  float* hs_u2i = (float*)alloc((size_t)NU * 256 * 4);
  float* hs_i2u = (float*)alloc((size_t)NI * 256 * 4);
  unsigned short* us_hu1 = (unsigned short*)alloc((size_t)NU * 256 * 2);
  unsigned short* us_hi1 = (unsigned short*)alloc((size_t)NI * 256 * 2);
  float* hs1_u2i = (float*)alloc((size_t)NU * 32 * 4);
  float* hs1_i2u = (float*)alloc((size_t)NI * 32 * 4);
  float* accb  = (float*)alloc((size_t)NU * 256 * 4);
  float* alpha = (float*)alloc((size_t)NE * 4 * 4);
  float* ealph = (float*)alloc((size_t)NE * 4 * 4);
  float* mbuf  = (float*)alloc((size_t)NU * 4 * 4);
  float* dbuf  = (float*)alloc((size_t)NU * 4 * 4);
  float* als   = (float*)alloc((size_t)NU * 4 * 4);
  float* ald   = (float*)alloc((size_t)NU * 4 * 4);
  unsigned short* wt_pu  = (unsigned short*)alloc(128 * 64 * 2);
  unsigned short* wt_pi  = (unsigned short*)alloc(128 * 64 * 2);
  unsigned short* wt_w0u = (unsigned short*)alloc(64 * 256 * 2);
  unsigned short* wt_w0i = (unsigned short*)alloc(64 * 256 * 2);
  unsigned short* wt_w1u = (unsigned short*)alloc(256 * 32 * 2);
  unsigned short* wt_w1i = (unsigned short*)alloc(256 * 32 * 2);
  float* us0u = (float*)alloc(64 * 4 * 4);
  float* ud0u = (float*)alloc(64 * 4 * 4);
  float* us0i = (float*)alloc(64 * 4 * 4);
  float* ud0i = (float*)alloc(64 * 4 * 4);
  float* us1u = (float*)alloc(256 * 4);
  float* ud1u = (float*)alloc(256 * 4);
  float* us1i = (float*)alloc(256 * 4);
  float* ud1i = (float*)alloc(256 * 4);

  const int T = 256;
#define G(n) (((n) + T - 1) / T), T, 0, stream
  auto gemm_blocks = [](int M, int N, int NT) { return ((M / 16) * (N / (16 * NT)) + 7) / 8; };

  // ---- convert inputs / weights to bf16 (weights transposed for B-fragment loads)
  k_f32_to_bf16<<<G(NU * 128)>>>(x_user, us_xu, NU * 128);
  k_f32_to_bf16<<<G(NI * 128)>>>(x_item, us_xi, NI * 128);
  k_transpose_bf16<<<G(128 * 64)>>>(p_user_w, wt_pu, 128, 64);
  k_transpose_bf16<<<G(128 * 64)>>>(p_item_w, wt_pi, 128, 64);
  k_transpose_bf16<<<G(64 * 256)>>>(l0u_ws, wt_w0u, 64, 256);
  k_transpose_bf16<<<G(64 * 256)>>>(l0i_ws, wt_w0i, 64, 256);
  k_transpose_bf16<<<G(256 * 32)>>>(l1u_ws, wt_w1u, 256, 32);
  k_transpose_bf16<<<G(256 * 32)>>>(l1i_ws, wt_w1i, 256, 32);
  // attention logit vectors: al = x @ (W @ a^T)
  k_make_u<<<G(64 * 4)>>>(l0u_ws, l0u_as, us0u, 64, 4, 64);
  k_make_u<<<G(64 * 4)>>>(l0u_wd, l0u_ad, ud0u, 64, 4, 64);
  k_make_u<<<G(64 * 4)>>>(l0i_ws, l0i_as, us0i, 64, 4, 64);
  k_make_u<<<G(64 * 4)>>>(l0i_wd, l0i_ad, ud0i, 64, 4, 64);
  k_make_u<<<G(256)>>>(l1u_ws, l1u_as, us1u, 256, 1, 32);
  k_make_u<<<G(256)>>>(l1u_wd, l1u_ad, ud1u, 256, 1, 32);
  k_make_u<<<G(256)>>>(l1i_ws, l1i_as, us1i, 256, 1, 32);
  k_make_u<<<G(256)>>>(l1i_wd, l1i_ad, ud1i, 256, 1, 32);

  // ---- projections: hu/hi = elu(x @ Pw + b), stored bf16
  k_wmma_gemm<2><<<gemm_blocks(NU, 64, 2), 256, 0, stream>>>(us_xu, wt_pu, p_user_b, nullptr, us_hu, NU, 64, 128, 1);
  k_wmma_gemm<2><<<gemm_blocks(NI, 64, 2), 256, 0, stream>>>(us_xi, wt_pi, p_item_b, nullptr, us_hi, NI, 64, 128, 1);

  // ---- layer-0 source transforms hs = h @ ws (f32 out; messages read these)
  k_wmma_gemm<2><<<gemm_blocks(NU, 256, 2), 256, 0, stream>>>(us_hu, wt_w0u, nullptr, hs_u2i, nullptr, NU, 256, 64, 0);
  k_wmma_gemm<2><<<gemm_blocks(NI, 256, 2), 256, 0, stream>>>(us_hi, wt_w0i, nullptr, hs_i2u, nullptr, NI, 256, 64, 0);

  const int *src_u2i = e_u2i, *dst_u2i = e_u2i + NE;
  const int *src_i2u = e_i2u, *dst_i2u = e_i2u + NE;

  // ---- layer-0 u2i: users -> items, out = hi1
  k_al<<<G(NU * 4)>>>(us_hu, us0u, als, NU, 64, 4);
  k_al<<<G(NI * 4)>>>(us_hi, ud0u, ald, NI, 64, 4);
  k_fill_f32<<<G(NI * 4)>>>(mbuf, -1e30f, NI * 4);
  k_fill_f32<<<G(NI * 4)>>>(dbuf, 0.f, NI * 4);
  k_fill_f32<<<G(NI * 256)>>>(accb, 0.f, NI * 256);
  k_alpha_max<<<G(NE * 4)>>>(src_u2i, dst_u2i, als, ald, alpha, mbuf, NE, 4);
  k_exp_sum<<<G(NE * 4)>>>(dst_u2i, alpha, mbuf, ealph, dbuf, NE, 4);
  k_scatter<<<G(NE * 4 * 16)>>>(src_u2i, dst_u2i, hs_u2i, ealph, dbuf, accb, NE, 4, 64);
  k_finish_elu_bf16<<<G(NI * 256)>>>(accb, l0u_b, us_hi1, NI * 256, 256);

  // ---- layer-0 i2u: items -> users, out = hu1
  k_al<<<G(NI * 4)>>>(us_hi, us0i, als, NI, 64, 4);
  k_al<<<G(NU * 4)>>>(us_hu, ud0i, ald, NU, 64, 4);
  k_fill_f32<<<G(NU * 4)>>>(mbuf, -1e30f, NU * 4);
  k_fill_f32<<<G(NU * 4)>>>(dbuf, 0.f, NU * 4);
  k_fill_f32<<<G(NU * 256)>>>(accb, 0.f, NU * 256);
  k_alpha_max<<<G(NE * 4)>>>(src_i2u, dst_i2u, als, ald, alpha, mbuf, NE, 4);
  k_exp_sum<<<G(NE * 4)>>>(dst_i2u, alpha, mbuf, ealph, dbuf, NE, 4);
  k_scatter<<<G(NE * 4 * 16)>>>(src_i2u, dst_i2u, hs_i2u, ealph, dbuf, accb, NE, 4, 64);
  k_finish_elu_bf16<<<G(NU * 256)>>>(accb, l0i_b, us_hu1, NU * 256, 256);

  // ---- layer-1 source transforms
  k_wmma_gemm<2><<<gemm_blocks(NU, 32, 2), 256, 0, stream>>>(us_hu1, wt_w1u, nullptr, hs1_u2i, nullptr, NU, 32, 256, 0);
  k_wmma_gemm<2><<<gemm_blocks(NI, 32, 2), 256, 0, stream>>>(us_hi1, wt_w1i, nullptr, hs1_i2u, nullptr, NI, 32, 256, 0);

  float* hu2 = (float*)d_out;            // first output
  float* hi2 = (float*)d_out + (size_t)NU * 32;  // second output

  // ---- layer-1 u2i: users -> items, out = hi2
  k_al<<<G(NU)>>>(us_hu1, us1u, als, NU, 256, 1);
  k_al<<<G(NI)>>>(us_hi1, ud1u, ald, NI, 256, 1);
  k_fill_f32<<<G(NI)>>>(mbuf, -1e30f, NI);
  k_fill_f32<<<G(NI)>>>(dbuf, 0.f, NI);
  k_fill_f32<<<G(NI * 32)>>>(hi2, 0.f, NI * 32);
  k_alpha_max<<<G(NE)>>>(src_u2i, dst_u2i, als, ald, alpha, mbuf, NE, 1);
  k_exp_sum<<<G(NE)>>>(dst_u2i, alpha, mbuf, ealph, dbuf, NE, 1);
  k_scatter<<<G(NE * 8)>>>(src_u2i, dst_u2i, hs1_u2i, ealph, dbuf, hi2, NE, 1, 32);
  k_add_bias<<<G(NI * 32)>>>(hi2, l1u_b, NI * 32, 32);

  // ---- layer-1 i2u: items -> users, out = hu2
  k_al<<<G(NI)>>>(us_hi1, us1i, als, NI, 256, 1);
  k_al<<<G(NU)>>>(us_hu1, ud1i, ald, NU, 256, 1);
  k_fill_f32<<<G(NU)>>>(mbuf, -1e30f, NU);
  k_fill_f32<<<G(NU)>>>(dbuf, 0.f, NU);
  k_fill_f32<<<G(NU * 32)>>>(hu2, 0.f, NU * 32);
  k_alpha_max<<<G(NE)>>>(src_i2u, dst_i2u, als, ald, alpha, mbuf, NE, 1);
  k_exp_sum<<<G(NE)>>>(dst_i2u, alpha, mbuf, ealph, dbuf, NE, 1);
  k_scatter<<<G(NE * 8)>>>(src_i2u, dst_i2u, hs1_i2u, ealph, dbuf, hu2, NE, 1, 32);
  k_add_bias<<<G(NU * 32)>>>(hu2, l1i_b, NU * 32, 32);
#undef G
  (void)in_sizes; (void)n_in; (void)out_size; (void)ws_size;
}
